// MixTransformer_50508815401287
// MI455X (gfx1250) — matrix operand order repaired and checked by reading, not compiled, not run on hardware
//
#include <hip/hip_runtime.h>

// ---------------- problem constants ----------------
#define B_   2
#define S_   1024
#define D_   2048
#define H_   16
#define HKV_ 8
#define HD_  128
#define F_   5632
#define E_   8
#define R_   16
#define M_   (B_*S_)

// ---------------- CDNA5 WMMA types ----------------
typedef __attribute__((ext_vector_type(16))) __bf16 v16bf;
typedef __attribute__((ext_vector_type(8)))  float  v8f;

union ABf {                 // one WMMA bf16 operand fragment (8 VGPRs)
  v16bf v;
  uint4 q[2];
  unsigned short s[16];
};

__device__ __forceinline__ unsigned short f2bf(float f) {
  unsigned int u = __float_as_uint(f);
  return (unsigned short)((u + 0x7FFFu + ((u >> 16) & 1u)) >> 16);
}

__device__ __forceinline__ v8f wmma_bf16(v16bf a, v16bf b, v8f c) {
  return __builtin_amdgcn_wmma_f32_16x16x32_bf16(false, a, false, b, (short)0, c,
                                                 false, false);
}

// ---------------- f32 -> bf16 elementwise ----------------
__global__ void cvt_f2bf_kernel(const float* __restrict__ x,
                                unsigned short* __restrict__ y, size_t n) {
  size_t i = (size_t)blockIdx.x * blockDim.x + threadIdx.x;
  if (i < n) y[i] = f2bf(x[i]);
}

// ---- tiled transpose: in[N,K] f32 -> out[K,N] bf16 (N,K multiples of 32) --
__global__ void transpose_f2bf_tiled(const float* __restrict__ in,
                                     unsigned short* __restrict__ out,
                                     int N, int K) {
  __shared__ float tile[32][33];
  int k0 = blockIdx.x * 32, n0 = blockIdx.y * 32;
  int tx = threadIdx.x, ty = threadIdx.y;  // block (32, 8)
#pragma unroll
  for (int i = 0; i < 32; i += 8)
    tile[ty + i][tx] = in[(size_t)(n0 + ty + i) * K + k0 + tx];  // tile[n][k]
  __syncthreads();
#pragma unroll
  for (int i = 0; i < 32; i += 8)
    out[(size_t)(k0 + ty + i) * N + n0 + tx] = f2bf(tile[tx][ty + i]);
}

// ---- simple transpose for small (rank-16) matrices: in[N,K] -> out[K,N] ---
__global__ void transpose_f2bf_simple(const float* __restrict__ in,
                                      unsigned short* __restrict__ out,
                                      int N, int K) {
  size_t i = (size_t)blockIdx.x * blockDim.x + threadIdx.x;
  if (i >= (size_t)N * K) return;
  int n = (int)(i % N);
  int k = (int)(i / N);
  out[i] = f2bf(in[(size_t)n * K + k]);
}

// ---------------- RMSNorm (one block per token, 256 thr) ----------------
__global__ void rmsnorm_kernel(const float* __restrict__ x,
                               const float* __restrict__ w,
                               unsigned short* __restrict__ yb,
                               float* __restrict__ yf) {
  int m = blockIdx.x;
  const float* row = x + (size_t)m * D_;
  float ss = 0.f;
  for (int d = threadIdx.x; d < D_; d += 256) { float v = row[d]; ss += v * v; }
  __shared__ float red[256];
  red[threadIdx.x] = ss; __syncthreads();
  for (int s = 128; s > 0; s >>= 1) {
    if (threadIdx.x < s) red[threadIdx.x] += red[threadIdx.x + s];
    __syncthreads();
  }
  float inv = rsqrtf(red[0] / (float)D_ + 1e-5f);
  for (int d = threadIdx.x; d < D_; d += 256) {
    float v = row[d] * inv * w[d];
    yb[(size_t)m * D_ + d] = f2bf(v);
    if (yf) yf[(size_t)m * D_ + d] = v;
  }
}

// ---------------- WMMA GEMM: C[M,N] = A[M,K](bf16) @ W^T, W given K-major --
// WT is [K,N] bf16. One wave computes a 16 x (NT*16) strip; NT is a template
// parameter so accumulators live in fixed VGPRs (no v_movrel indexing).
template <int NT>
__global__ void gemm_wmma(const unsigned short* __restrict__ A,
                          const unsigned short* __restrict__ WT,
                          float* __restrict__ C,
                          const float* __restrict__ resid,
                          unsigned short* __restrict__ Cbf,
                          int M, int N, int K, int accum) {
  int wave = threadIdx.x >> 5;
  int lane = threadIdx.x & 31;
  int lm = lane & 15, half = lane >> 4;
  int nt = N >> 4;                 // 16-wide N tiles
  int ntg = nt / NT;               // strip groups (nt % NT == 0 by dispatch)
  int g = blockIdx.x * 8 + wave;
  if (g >= (M >> 4) * ntg) return;
  int tm  = (g / ntg) << 4;
  int tn0 = (g % ntg) * NT;        // first tile index in strip

  v8f acc[NT];
#pragma unroll
  for (int j = 0; j < NT; ++j)
#pragma unroll
    for (int r = 0; r < 8; ++r) acc[j][r] = 0.f;
  if (accum) {
#pragma unroll
    for (int j = 0; j < NT; ++j)
#pragma unroll
      for (int r = 0; r < 8; ++r)
        acc[j][r] = C[(size_t)(tm + r + half * 8) * N + ((tn0 + j) << 4) + lm];
  }

  const unsigned short* arow = A + (size_t)(tm + lm) * K;
  for (int k0 = 0; k0 < K; k0 += 32) {
    __builtin_prefetch(arow + k0 + 256, 0, 3);   // gfx1250 global_prefetch_b8
    ABf a;
    a.q[0] = *(const uint4*)(arow + k0 + half * 8);
    a.q[1] = *(const uint4*)(arow + k0 + half * 8 + 16);
    const unsigned short* brow = WT + (size_t)(k0 + half * 8) * N + (tn0 << 4) + lm;
#pragma unroll
    for (int j = 0; j < NT; ++j) {
      ABf b;
#pragma unroll
      for (int s = 0; s < 16; ++s)
        b.s[s] = brow[(size_t)((s & 7) + (s >> 3) * 16) * N + j * 16];
      acc[j] = wmma_bf16(a.v, b.v, acc[j]);
    }
  }

#pragma unroll
  for (int j = 0; j < NT; ++j)
#pragma unroll
    for (int r = 0; r < 8; ++r) {
      size_t idx = (size_t)(tm + r + half * 8) * N + ((tn0 + j) << 4) + lm;
      float v = acc[j][r];
      if (resid) v += resid[idx];
      C[idx] = v;
      if (Cbf) Cbf[idx] = f2bf(v);
    }
}

// ---------------- RoPE for Q: f32 [M, H*HD] -> bf16 same layout -----------
__global__ void rope_q_kernel(const float* __restrict__ x,
                              const float* __restrict__ cosT,
                              const float* __restrict__ sinT,
                              unsigned short* __restrict__ yb) {
  size_t idx = (size_t)blockIdx.x * blockDim.x + threadIdx.x;
  size_t total = (size_t)M_ * H_ * (HD_ / 2);
  if (idx >= total) return;
  int i = (int)(idx % (HD_ / 2));
  int h = (int)((idx / (HD_ / 2)) % H_);
  size_t m = idx / ((size_t)(HD_ / 2) * H_);
  int s = (int)(m % S_);
  float c = cosT[(size_t)s * (HD_ / 2) + i];
  float sn = sinT[(size_t)s * (HD_ / 2) + i];
  size_t base = m * (size_t)(H_ * HD_) + (size_t)h * HD_ + 2 * i;
  float x1 = x[base], x2 = x[base + 1];
  yb[base]     = f2bf(x1 * c - x2 * sn);
  yb[base + 1] = f2bf(x1 * sn + x2 * c);
}

// ---- RoPE for K: f32 [M, HKV*HD] -> bf16 TRANSPOSED [B, HKV, HD, S] ------
// (head-dim-major so QK^T B-fragment lanes read consecutive S (16 loads of stride S))
__global__ void rope_kt_kernel(const float* __restrict__ x,
                               const float* __restrict__ cosT,
                               const float* __restrict__ sinT,
                               unsigned short* __restrict__ yt) {
  size_t idx = (size_t)blockIdx.x * blockDim.x + threadIdx.x;
  size_t total = (size_t)M_ * HKV_ * (HD_ / 2);
  if (idx >= total) return;
  int i = (int)(idx % (HD_ / 2));
  int h = (int)((idx / (HD_ / 2)) % HKV_);
  size_t m = idx / ((size_t)(HD_ / 2) * HKV_);
  int s = (int)(m % S_);
  int b = (int)(m / S_);
  float c = cosT[(size_t)s * (HD_ / 2) + i];
  float sn = sinT[(size_t)s * (HD_ / 2) + i];
  size_t base = m * (size_t)(HKV_ * HD_) + (size_t)h * HD_ + 2 * i;
  float x1 = x[base], x2 = x[base + 1];
  size_t obase = ((size_t)(b * HKV_ + h) * HD_ + 2 * i) * S_ + s;
  yt[obase]      = f2bf(x1 * c - x2 * sn);
  yt[obase + S_] = f2bf(x1 * sn + x2 * c);
}

// ---------------- flash attention, one wave = (b,h,16-query tile) ---------
// Q tile staged in LDS (wave-private) so the loop-invariant A fragments do
// NOT occupy VGPRs or spill to scratch: in-loop they are two ds_load_b128.
__global__ void __launch_bounds__(256)
attn_kernel(const unsigned short* __restrict__ qb,
            const unsigned short* __restrict__ kt,  // [B,HKV,HD,S]
            const unsigned short* __restrict__ vb,  // [B,S,HKV,HD]
            const float* __restrict__ mask,
            unsigned short* __restrict__ ob) {
  __shared__ unsigned short qlds[8][16][HD_];  // 32 KB: per-wave 16x128 Q tile
  __shared__ unsigned short plds[8][16][32];   //  8 KB: per-wave P staging
  int wave = threadIdx.x >> 5, lane = threadIdx.x & 31;
  int lm = lane & 15, half = lane >> 4;
  int task = blockIdx.x * 8 + wave;            // 2048 tasks exactly
  int bb = task / (H_ * (S_ / 16));
  int h  = (task / (S_ / 16)) % H_;
  int q0 = (task % (S_ / 16)) * 16;
  int kvh = h / (H_ / HKV_);

  // stage this wave's Q tile: 16 rows x 128 cols bf16 = 256 x 16B chunks
#pragma unroll
  for (int c = 0; c < 8; ++c) {
    int chunk = c * 32 + lane;          // 0..255
    int row = chunk >> 4;
    int col = (chunk & 15) << 3;
    *(uint4*)&qlds[wave][row][col] =
        *(const uint4*)(qb + ((size_t)(bb * S_ + q0 + row) * H_ + h) * HD_ + col);
  }

  v8f o[8];
  float rmax[8], rsum[8];
#pragma unroll
  for (int r = 0; r < 8; ++r) {
    rmax[r] = -1e30f; rsum[r] = 0.f;
#pragma unroll
    for (int t = 0; t < 8; ++t) o[t][r] = 0.f;
  }

  const unsigned short* qrow_l = &qlds[wave][lm][0];   // LDS row for A frag
  const unsigned short* kbase = kt + (size_t)(bb * HKV_ + kvh) * HD_ * S_;
  const float inv_sqrt_hd = 0.08838834764831845f;  // 1/sqrt(128)

  for (int n0 = 0; n0 < S_; n0 += 32) {
    v8f c0, c1;
#pragma unroll
    for (int r = 0; r < 8; ++r) { c0[r] = 0.f; c1[r] = 0.f; }
#pragma unroll
    for (int kc = 0; kc < 4; ++kc) {
      ABf a;
      int koff = kc * 32 + half * 8;
      a.q[0] = *(const uint4*)(qrow_l + koff);        // ds_load_b128
      a.q[1] = *(const uint4*)(qrow_l + koff + 16);   // ds_load_b128
      const unsigned short* krow = kbase + (size_t)(kc * 32 + half * 8) * S_ + n0 + lm;
      ABf b0, b1;
#pragma unroll
      for (int s = 0; s < 16; ++s) {
        b0.s[s] = krow[(size_t)((s & 7) + (s >> 3) * 16) * S_];
        b1.s[s] = krow[(size_t)((s & 7) + (s >> 3) * 16) * S_ + 16];
      }
      c0 = wmma_bf16(a.v, b0.v, c0);
      c1 = wmma_bf16(a.v, b1.v, c1);
    }
    // scale + mask + online softmax (rows: VGPR r -> M = r + 8*half)
#pragma unroll
    for (int r = 0; r < 8; ++r) {
      int row = q0 + r + half * 8;
      c0[r] = c0[r] * inv_sqrt_hd + mask[(size_t)row * S_ + n0 + lm];
      c1[r] = c1[r] * inv_sqrt_hd + mask[(size_t)row * S_ + n0 + 16 + lm];
      float tm = fmaxf(c0[r], c1[r]);
      for (int xm = 1; xm < 16; xm <<= 1) tm = fmaxf(tm, __shfl_xor(tm, xm, 32));
      float nm = fmaxf(rmax[r], tm);
      float scl = __expf(rmax[r] - nm);
      float p0 = __expf(c0[r] - nm), p1 = __expf(c1[r] - nm);
      float ts = p0 + p1;
      for (int xm = 1; xm < 16; xm <<= 1) ts += __shfl_xor(ts, xm, 32);
      rsum[r] = rsum[r] * scl + ts;
      rmax[r] = nm;
#pragma unroll
      for (int t = 0; t < 8; ++t) o[t][r] *= scl;
      int mrow = r + half * 8;
      plds[wave][mrow][lm]      = f2bf(p0);
      plds[wave][mrow][lm + 16] = f2bf(p1);
    }
    // reload P in A-fragment layout (same-wave LDS ops are in order)
    ABf ap;
    const unsigned short* prow = &plds[wave][lm][0];
    ap.q[0] = *(const uint4*)(prow + half * 8);
    ap.q[1] = *(const uint4*)(prow + half * 8 + 16);
    // P(16x32) @ V(32x128): 8 hd tiles; V elements contiguous along HD
    const unsigned short* vrow =
        vb + ((size_t)(bb * S_ + n0 + half * 8) * HKV_ + kvh) * HD_ + lm;
#pragma unroll
    for (int t = 0; t < 8; ++t) {
      ABf bv;
#pragma unroll
      for (int s = 0; s < 16; ++s)
        bv.s[s] = vrow[(size_t)((s & 7) + (s >> 3) * 16) * HKV_ * HD_ + t * 16];
      o[t] = wmma_bf16(ap.v, bv.v, o[t]);
    }
  }
  // epilogue: normalize, store bf16 [B,S,H*HD]
#pragma unroll
  for (int r = 0; r < 8; ++r) {
    float inv = 1.f / rsum[r];
    int row = q0 + r + half * 8;
#pragma unroll
    for (int t = 0; t < 8; ++t)
      ob[((size_t)(bb * S_ + row)) * (H_ * HD_) + h * HD_ + t * 16 + lm] =
          f2bf(o[t][r] * inv);
  }
}

// ---------------- MoE gating (matches reference's softmax over S axis) ----
__global__ void gate_logits_kernel(const float* __restrict__ sn,
                                   const float* __restrict__ gw,
                                   float* __restrict__ logits) {
  int m = blockIdx.x;
  float p[E_];
#pragma unroll
  for (int e = 0; e < E_; ++e) p[e] = 0.f;
  for (int k = threadIdx.x; k < D_; k += 256) {
    float s = sn[(size_t)m * D_ + k];
#pragma unroll
    for (int e = 0; e < E_; ++e) p[e] += s * gw[(size_t)e * D_ + k];
  }
  __shared__ float red[256];
  for (int e = 0; e < E_; ++e) {
    red[threadIdx.x] = p[e]; __syncthreads();
    for (int s = 128; s > 0; s >>= 1) {
      if (threadIdx.x < s) red[threadIdx.x] += red[threadIdx.x + s];
      __syncthreads();
    }
    if (threadIdx.x == 0) logits[(size_t)m * E_ + e] = red[0];
    __syncthreads();
  }
}

__global__ void gate_z_kernel(const float* __restrict__ logits,
                              float* __restrict__ Z) {
  int b = blockIdx.x / E_, e = blockIdx.x % E_;
  float s = 0.f;
  for (int t = threadIdx.x; t < S_; t += 256)
    s += __expf(logits[((size_t)b * S_ + t) * E_ + e]);
  __shared__ float red[256];
  red[threadIdx.x] = s; __syncthreads();
  for (int k = 128; k > 0; k >>= 1) {
    if (threadIdx.x < k) red[threadIdx.x] += red[threadIdx.x + k];
    __syncthreads();
  }
  if (threadIdx.x == 0) Z[blockIdx.x] = red[0];
}

__global__ void gate_topk_kernel(const float* __restrict__ logits,
                                 const float* __restrict__ Z,
                                 float* __restrict__ rwfull) {
  int m = blockIdx.x * blockDim.x + threadIdx.x;
  if (m >= M_) return;
  int b = m / S_;
  float rw[E_];
#pragma unroll
  for (int e = 0; e < E_; ++e)
    rw[e] = __expf(logits[(size_t)m * E_ + e]) / Z[b * E_ + e];
  int i1 = 0;
  for (int e = 1; e < E_; ++e) if (rw[e] > rw[i1]) i1 = e;
  int i2 = -1;
  for (int e = 0; e < E_; ++e) {
    if (e == i1) continue;
    if (i2 < 0 || rw[e] > rw[i2]) i2 = e;
  }
  float s = rw[i1] + rw[i2];
#pragma unroll
  for (int e = 0; e < E_; ++e)
    rwfull[(size_t)m * E_ + e] =
        (e == i1) ? rw[i1] / s : ((e == i2) ? rw[i2] / s : 0.f);
}

// ---- fused per-expert: rank-16 LoRA delta, SiLU gate, weighted accum -----
__global__ void expert_sr_kernel(const float* __restrict__ c1,
                                 const float* __restrict__ c3,
                                 const float* __restrict__ t1,
                                 const float* __restrict__ t3,
                                 const float* __restrict__ w1b,  // [F,R]
                                 const float* __restrict__ w3b,  // [F,R]
                                 const float* __restrict__ rwfull,
                                 int e, int first,
                                 unsigned short* __restrict__ srb,
                                 float* __restrict__ srw) {
  int m = blockIdx.y;
  __shared__ float s1[R_], s3[R_];
  if (threadIdx.x < R_) {
    s1[threadIdx.x] = t1[(size_t)m * R_ + threadIdx.x];
    s3[threadIdx.x] = t3[(size_t)m * R_ + threadIdx.x];
  }
  __syncthreads();
  int f = blockIdx.x * blockDim.x + threadIdx.x;
  if (f >= F_) return;
  float d1 = 0.f, d3 = 0.f;
#pragma unroll
  for (int r = 0; r < R_; ++r) {
    d1 += s1[r] * w1b[(size_t)f * R_ + r];
    d3 += s3[r] * w3b[(size_t)f * R_ + r];
  }
  size_t idx = (size_t)m * F_ + f;
  float h1 = c1[idx] + 2.f * d1;
  float h3 = c3[idx] + 2.f * d3;
  float sr = (h1 / (1.f + __expf(-h1))) * h3;   // silu(h1)*h3
  srb[idx] = f2bf(sr);
  float a = rwfull[(size_t)m * E_ + e] * sr;
  srw[idx] = first ? a : srw[idx] + a;
}

__global__ void uw_pack_kernel(const float* __restrict__ u,
                               const float* __restrict__ rwfull, int e,
                               unsigned short* __restrict__ Uw) {
  int i = blockIdx.x * blockDim.x + threadIdx.x;  // M*R
  if (i >= M_ * R_) return;
  int m = i >> 4, r = i & 15;
  Uw[(size_t)m * (E_ * R_) + e * R_ + r] =
      f2bf(2.f * rwfull[(size_t)m * E_ + e] * u[i]);
}

// w2_b: [E, D, R] -> bf16 K-major [E*R, D]
__global__ void w2b_pack_kernel(const float* __restrict__ w2b,
                                unsigned short* __restrict__ out) {
  int i = blockIdx.x * blockDim.x + threadIdx.x;
  if (i >= D_ * E_ * R_) return;
  int k = i / D_;            // k = e*R + r
  int n = i % D_;
  int e = k >> 4, r = k & 15;
  out[i] = f2bf(w2b[((size_t)e * D_ + n) * R_ + r]);
}

// =========================== host side ====================================
extern "C" void kernel_launch(void* const* d_in, const int* in_sizes, int n_in,
                              void* d_out, int out_size, void* d_ws,
                              size_t ws_size, hipStream_t stream) {
  (void)in_sizes; (void)n_in; (void)out_size; (void)ws_size;
  const float* data  = (const float*)d_in[0];
  const float* mask  = (const float*)d_in[1];
  const float* rcos  = (const float*)d_in[2];
  const float* rsin  = (const float*)d_in[3];
  const float* anw   = (const float*)d_in[4];
  const float* fnw   = (const float*)d_in[5];
  const float* wq    = (const float*)d_in[6];
  const float* wk    = (const float*)d_in[7];
  const float* wv    = (const float*)d_in[8];
  const float* wo    = (const float*)d_in[9];
  const float* gw    = (const float*)d_in[10];
  const float* w1    = (const float*)d_in[11];
  const float* w2    = (const float*)d_in[12];
  const float* w3    = (const float*)d_in[13];
  const float* w1a   = (const float*)d_in[14];
  const float* w1b   = (const float*)d_in[15];
  const float* w3a   = (const float*)d_in[16];
  const float* w3b   = (const float*)d_in[17];
  const float* w2a   = (const float*)d_in[18];
  const float* w2b   = (const float*)d_in[19];
  float* out = (float*)d_out;

  // bump allocator on workspace
  char* ws = (char*)d_ws;
  size_t off = 0;
  auto alloc = [&](size_t bytes) -> void* {
    off = (off + 255) & ~(size_t)255;
    void* p = ws + off;
    off += bytes;
    return p;
  };
  auto aBF = [&](size_t n) { return (unsigned short*)alloc(n * 2); };
  auto aF  = [&](size_t n) { return (float*)alloc(n * 4); };

  // bf16 activation buffers
  unsigned short* xnb   = aBF((size_t)M_ * D_);
  unsigned short* snb   = aBF((size_t)M_ * D_);
  unsigned short* qbf   = aBF((size_t)M_ * H_ * HD_);
  unsigned short* ktb   = aBF((size_t)B_ * HKV_ * HD_ * S_);
  unsigned short* vbf   = aBF((size_t)M_ * HKV_ * HD_);
  unsigned short* attnb = aBF((size_t)M_ * H_ * HD_);
  unsigned short* srb   = aBF((size_t)M_ * F_);
  unsigned short* srwb  = aBF((size_t)M_ * F_);
  unsigned short* Uwb   = aBF((size_t)M_ * E_ * R_);
  // K-major bf16 weights
  unsigned short* wqT   = aBF((size_t)D_ * D_);
  unsigned short* wkT   = aBF((size_t)D_ * HKV_ * HD_);
  unsigned short* wvT   = aBF((size_t)D_ * HKV_ * HD_);
  unsigned short* woT   = aBF((size_t)D_ * D_);
  unsigned short* w1T   = aBF((size_t)D_ * F_);
  unsigned short* w3T   = aBF((size_t)D_ * F_);
  unsigned short* w2T   = aBF((size_t)F_ * D_);
  unsigned short* w1aT  = aBF((size_t)E_ * D_ * R_);
  unsigned short* w3aT  = aBF((size_t)E_ * D_ * R_);
  unsigned short* w2aT  = aBF((size_t)E_ * F_ * R_);
  unsigned short* w2bT  = aBF((size_t)E_ * R_ * D_);
  // f32 buffers
  float* qf     = aF((size_t)M_ * H_ * HD_);
  float* kf     = aF((size_t)M_ * HKV_ * HD_);
  float* vf     = aF((size_t)M_ * HKV_ * HD_);
  float* snf    = aF((size_t)M_ * D_);
  float* residf = aF((size_t)M_ * D_);
  float* c1f    = aF((size_t)M_ * F_);
  float* c3f    = aF((size_t)M_ * F_);
  float* srwf   = aF((size_t)M_ * F_);
  float* t1f    = aF((size_t)M_ * R_);
  float* t3f    = aF((size_t)M_ * R_);
  float* uf     = aF((size_t)M_ * R_);
  float* logits = aF((size_t)M_ * E_);
  float* Zf     = aF((size_t)B_ * E_);
  float* rwfull = aF((size_t)M_ * E_);

  auto transposeBig = [&](const float* src, unsigned short* dst, int N, int K) {
    transpose_f2bf_tiled<<<dim3(K / 32, N / 32), dim3(32, 8), 0, stream>>>(
        src, dst, N, K);
  };
  auto transposeSmall = [&](const float* src, unsigned short* dst, int N, int K) {
    size_t n = (size_t)N * K;
    transpose_f2bf_simple<<<(unsigned)((n + 255) / 256), 256, 0, stream>>>(
        src, dst, N, K);
  };
  auto gemm = [&](const unsigned short* A, const unsigned short* WT, float* C,
                  const float* resid, unsigned short* Cbf, int M, int N, int K,
                  int accum) {
    int nt = N / 16;
    if (nt % 4 == 0) {
      int groups = (M / 16) * (nt / 4);
      gemm_wmma<4><<<(groups + 7) / 8, 256, 0, stream>>>(A, WT, C, resid, Cbf,
                                                         M, N, K, accum);
    } else {
      int groups = (M / 16) * nt;
      gemm_wmma<1><<<(groups + 7) / 8, 256, 0, stream>>>(A, WT, C, resid, Cbf,
                                                         M, N, K, accum);
    }
  };

  // ---- stage weights to K-major bf16 ----
  transposeBig(wq, wqT, D_, D_);
  transposeBig(wk, wkT, HKV_ * HD_, D_);
  transposeBig(wv, wvT, HKV_ * HD_, D_);
  transposeBig(wo, woT, D_, D_);
  transposeBig(w1, w1T, F_, D_);
  transposeBig(w3, w3T, F_, D_);
  transposeBig(w2, w2T, D_, F_);
  for (int e = 0; e < E_; ++e) {
    transposeSmall(w1a + (size_t)e * R_ * D_, w1aT + (size_t)e * D_ * R_, R_, D_);
    transposeSmall(w3a + (size_t)e * R_ * D_, w3aT + (size_t)e * D_ * R_, R_, D_);
    transposeSmall(w2a + (size_t)e * R_ * F_, w2aT + (size_t)e * F_ * R_, R_, F_);
  }
  w2b_pack_kernel<<<(D_ * E_ * R_ + 255) / 256, 256, 0, stream>>>(w2b, w2bT);

  // ---- attention ----
  rmsnorm_kernel<<<M_, 256, 0, stream>>>(data, anw, xnb, nullptr);
  gemm(xnb, wqT, qf, nullptr, nullptr, M_, H_ * HD_, D_, 0);
  gemm(xnb, wkT, kf, nullptr, nullptr, M_, HKV_ * HD_, D_, 0);
  gemm(xnb, wvT, vf, nullptr, vbf, M_, HKV_ * HD_, D_, 0);
  {
    size_t nq = (size_t)M_ * H_ * (HD_ / 2);
    rope_q_kernel<<<(unsigned)((nq + 255) / 256), 256, 0, stream>>>(qf, rcos,
                                                                    rsin, qbf);
    size_t nk = (size_t)M_ * HKV_ * (HD_ / 2);
    rope_kt_kernel<<<(unsigned)((nk + 255) / 256), 256, 0, stream>>>(kf, rcos,
                                                                     rsin, ktb);
  }
  attn_kernel<<<(B_ * H_ * (S_ / 16)) / 8, 256, 0, stream>>>(qbf, ktb, vbf, mask,
                                                             attnb);
  // data = data + attn @ wo^T
  gemm(attnb, woT, residf, data, nullptr, M_, D_, D_, 0);

  // ---- FFN / MoE ----
  rmsnorm_kernel<<<M_, 256, 0, stream>>>(residf, fnw, snb, snf);
  gate_logits_kernel<<<M_, 256, 0, stream>>>(snf, gw, logits);
  gate_z_kernel<<<B_ * E_, 256, 0, stream>>>(logits, Zf);
  gate_topk_kernel<<<(M_ + 255) / 256, 256, 0, stream>>>(logits, Zf, rwfull);

  gemm(snb, w1T, c1f, nullptr, nullptr, M_, F_, D_, 0);
  gemm(snb, w3T, c3f, nullptr, nullptr, M_, F_, D_, 0);

  for (int e = 0; e < E_; ++e) {
    gemm(snb, w1aT + (size_t)e * D_ * R_, t1f, nullptr, nullptr, M_, R_, D_, 0);
    gemm(snb, w3aT + (size_t)e * D_ * R_, t3f, nullptr, nullptr, M_, R_, D_, 0);
    expert_sr_kernel<<<dim3((F_ + 255) / 256, M_), 256, 0, stream>>>(
        c1f, c3f, t1f, t3f, w1b + (size_t)e * F_ * R_, w3b + (size_t)e * F_ * R_,
        rwfull, e, e == 0 ? 1 : 0, srb, srwf);
    gemm(srb, w2aT + (size_t)e * F_ * R_, uf, nullptr, nullptr, M_, R_, F_, 0);
    uw_pack_kernel<<<(M_ * R_ + 255) / 256, 256, 0, stream>>>(uf, rwfull, e, Uwb);
  }

  cvt_f2bf_kernel<<<(unsigned)(((size_t)M_ * F_ + 255) / 256), 256, 0, stream>>>(
      srwf, srwb, (size_t)M_ * F_);
  // out = resid + srw @ w2^T ; then out += Uw @ w2bcat^T
  gemm(srwb, w2T, out, residf, nullptr, M_, D_, F_, 0);
  gemm(Uwb, w2bT, out, nullptr, nullptr, M_, D_, E_ * R_, 1);
}
